// ChunkedAttention_50886772523164
// MI455X (gfx1250) — matrix operand order repaired and checked by reading, not compiled, not run on hardware
//
#include <hip/hip_runtime.h>
#include <hip/hip_bf16.h>

#define MASK_VAL (-50000.0f)

typedef __attribute__((ext_vector_type(16))) _Float16 v16h;
typedef __attribute__((ext_vector_type(8)))  _Float16 v8h;
typedef __attribute__((ext_vector_type(8)))  float    v8f;

#define WMMA_F16(a, b, c) \
  __builtin_amdgcn_wmma_f32_16x16x32_f16(false, (a), false, (b), (short)0, (c), false, false)

// ---------------------------------------------------------------------------
// A-fragment (16x32 f16, M x K) loader. ISA layout:
//   lanes 0-15 : row M = lane,    K halves {k0+0..7,  k0+16..23}
//   lanes 16-31: row M = lane-16, K halves {k0+8..15, k0+24..31}
// Two contiguous 16-byte chunks per lane.
// ---------------------------------------------------------------------------
__device__ __forceinline__ v16h load_a_frag(const _Float16* __restrict__ base,
                                            int lda, int row0, int k0) {
  int lane = threadIdx.x & 31;
  int row  = row0 + (lane & 15);
  int kb   = k0 + ((lane < 16) ? 0 : 8);
  const _Float16* p = base + (size_t)row * lda + kb;
  v8h lo = *(const v8h*)p;
  v8h hi = *(const v8h*)(p + 16);
  return __builtin_shufflevector(lo, hi, 0, 1, 2, 3, 4, 5, 6, 7,
                                 8, 9, 10, 11, 12, 13, 14, 15);
}

// ---------------------------------------------------------------------------
// B-fragment (32x16 f16, K x N) loader, where B[k][n] = W[col0+n][k0+k]
// (W row-major over k).  ISA layout:
//   lanes 0-15 : col N = lane,    K halves {k0+0..7,  k0+16..23}
//   lanes 16-31: col N = lane-16, K halves {k0+8..15, k0+24..31}
// Two contiguous 16-byte chunks per lane.
// ---------------------------------------------------------------------------
__device__ __forceinline__ v16h load_b_frag(const _Float16* __restrict__ base,
                                            int ldb, int col0, int k0) {
  int lane = threadIdx.x & 31;
  int col  = col0 + (lane & 15);
  int kb   = k0 + ((lane < 16) ? 0 : 8);
  const _Float16* p = base + (size_t)col * ldb + kb;
  v8h lo = *(const v8h*)p;
  v8h hi = *(const v8h*)(p + 16);
  return __builtin_shufflevector(lo, hi, 0, 1, 2, 3, 4, 5, 6, 7,
                                 8, 9, 10, 11, 12, 13, 14, 15);
}

// ---------------------------------------------------------------------------
// f32 -> f16 conversion
// ---------------------------------------------------------------------------
__global__ void cvt_f32_to_f16_kernel(const float* __restrict__ src,
                                      _Float16* __restrict__ dst, int n) {
  int i = blockIdx.x * 256 + threadIdx.x;
  if (i < n) dst[i] = (_Float16)src[i];
}

// ---------------------------------------------------------------------------
// GEMM: C[M,N] = A[M,K] * W[N,K]^T , f16 in, f32 accumulate.
// Block = 256 threads = 8 waves (2 along M x 4 along N).
// Wave tile 32x64 (8 wmma / k-step, two A-frags share each B-frag),
// block tile 64x256.
// mode 0: write f32 to outF[m*N + n] (scaled)
// mode 1: n <  split -> f16 head layout  [b,h,mi,dh]     (outH0, scaled)
//         n >= split -> f16 transposed   [b,h,dh,mi]     (outH1)
// ---------------------------------------------------------------------------
__global__ __launch_bounds__(256) void wmma_gemm_kernel(
    const _Float16* __restrict__ A, const _Float16* __restrict__ W,
    int M, int N, int K, int mode, float scale,
    _Float16* __restrict__ outH0, _Float16* __restrict__ outH1, int split,
    float* __restrict__ outF) {
  int wave = threadIdx.x >> 5;
  int lane = threadIdx.x & 31;
  int wm = wave & 1;   // 2 waves along M, 32 rows each
  int wn = wave >> 1;  // 4 waves along N, 64 cols each
  int m0 = blockIdx.x * 64 + wm * 32;
  int n0 = blockIdx.y * 256 + wn * 64;

  v8f acc[2][4];
#pragma unroll
  for (int r = 0; r < 2; ++r)
#pragma unroll
    for (int t = 0; t < 4; ++t) acc[r][t] = (v8f)0.0f;

  for (int k0 = 0; k0 < K; k0 += 32) {
    v16h a0 = load_a_frag(A, K, m0, k0);
    v16h a1 = load_a_frag(A, K, m0 + 16, k0);
#pragma unroll
    for (int t = 0; t < 4; ++t) {
      v16h b = load_b_frag(W, K, n0 + t * 16, k0);
      acc[0][t] = WMMA_F16(a0, b, acc[0][t]);
      acc[1][t] = WMMA_F16(a1, b, acc[1][t]);
    }
  }

  // Epilogue. C/D layout: VGPR i -> row i (lanes 0-15) / row i+8 (lanes 16-31),
  // col = lane & 15 within each 16-column tile.
  int rowoff = (lane < 16) ? 0 : 8;
  int colL   = lane & 15;
#pragma unroll
  for (int r = 0; r < 2; ++r) {
#pragma unroll
    for (int t = 0; t < 4; ++t) {
#pragma unroll
      for (int i = 0; i < 8; ++i) {
        int m = m0 + r * 16 + rowoff + i;
        int n = n0 + t * 16 + colL;
        float v = acc[r][t][i] * scale;
        if (mode == 0) {
          outF[(size_t)m * N + n] = v;
        } else {
          int b  = m >> 11;      // m / 2048
          int mi = m & 2047;
          if (n < split) {
            int h = n >> 6, c = n & 63;
            outH0[(((size_t)(b * 16 + h)) * 2048 + mi) * 64 + c] = (_Float16)v;
          } else {
            int o = n - split;
            int h = o >> 6, c = o & 63;
            outH1[(((size_t)(b * 16 + h)) * 64 + c) * 2048 + mi] = (_Float16)v;
          }
        }
      }
    }
  }
}

// ---------------------------------------------------------------------------
// Flash-style causal attention.  grid = (n/128, H, B), 256 threads = 8 waves.
// Wave w owns q rows [blk*128 + w*16, +16).  Uniform loop over 32-key blocks
// up to the causal frontier of the whole 128-row block; fully-masked tiles
// contribute exactly 0 (exp(-50000 - m) underflows), matching the reference's
// MASK_VAL softmax.
//   q,k : f16 [b,h,n,dh]   (q pre-scaled by dh^-0.5)
//   vT  : f16 [b,h,dh,n]   (transposed so PV B-fragments are contiguous)
//   out : f16 [b,n,d]      (heads merged, softmax-normalized)
// ---------------------------------------------------------------------------
__global__ __launch_bounds__(256) void attn_kernel(
    const _Float16* __restrict__ q, const _Float16* __restrict__ k,
    const _Float16* __restrict__ vT, _Float16* __restrict__ out) {
  __align__(32) __shared__ _Float16 sP[8][16][32];

  const int n = 2048, dh = 64;
  int wave = threadIdx.x >> 5;
  int lane = threadIdx.x & 31;
  int blk = blockIdx.x;
  int h   = blockIdx.y;
  int b   = blockIdx.z;

  const _Float16* qh = q  + ((size_t)(b * 16 + h)) * n * dh;
  const _Float16* kh = k  + ((size_t)(b * 16 + h)) * n * dh;
  const _Float16* vh = vT + ((size_t)(b * 16 + h)) * dh * n;

  int qbase  = blk * 128 + wave * 16;
  int rowoff = (lane < 16) ? 0 : 8;
  int colL   = lane & 15;

  // Q fragments (dh split into two K=32 steps)
  v16h qa0 = load_a_frag(qh, dh, qbase, 0);
  v16h qa1 = load_a_frag(qh, dh, qbase, 32);

  float mrow[8], lrow[8];
#pragma unroll
  for (int i = 0; i < 8; ++i) { mrow[i] = -1e30f; lrow[i] = 0.0f; }
  v8f o[4];
#pragma unroll
  for (int t = 0; t < 4; ++t) o[t] = (v8f)0.0f;

  int nblocks = (blk + 1) * 4;  // keys up to (blk+1)*128, 32 per block
  for (int kb = 0; kb < nblocks; ++kb) {
    int j0 = kb * 32;

    // S = Q * K^T  (two 16x16 column tiles, each dh-accumulated in 2 steps)
    v8f s0 = (v8f)0.0f, s1 = (v8f)0.0f;
    {
      v16h b00 = load_b_frag(kh, dh, j0, 0);
      v16h b01 = load_b_frag(kh, dh, j0, 32);
      s0 = WMMA_F16(qa0, b00, s0);
      s0 = WMMA_F16(qa1, b01, s0);
      v16h b10 = load_b_frag(kh, dh, j0 + 16, 0);
      v16h b11 = load_b_frag(kh, dh, j0 + 16, 32);
      s1 = WMMA_F16(qa0, b10, s1);
      s1 = WMMA_F16(qa1, b11, s1);
    }

    // Mask + online softmax stats (row = VGPR index within 16-lane group)
    float alpha[8];
    v8f p0, p1;
#pragma unroll
    for (int i = 0; i < 8; ++i) {
      int qrow = qbase + rowoff + i;
      float a0 = ((j0 + colL)      > qrow) ? MASK_VAL : s0[i];
      float a1 = ((j0 + 16 + colL) > qrow) ? MASK_VAL : s1[i];
      float mx = fmaxf(a0, a1);
#pragma unroll
      for (int d = 1; d < 16; d <<= 1) mx = fmaxf(mx, __shfl_xor(mx, d, 32));
      float mn = fmaxf(mrow[i], mx);
      float al = __expf(mrow[i] - mn);
      float e0 = __expf(a0 - mn);
      float e1 = __expf(a1 - mn);
      float sum = e0 + e1;
#pragma unroll
      for (int d = 1; d < 16; d <<= 1) sum += __shfl_xor(sum, d, 32);
      lrow[i]  = lrow[i] * al + sum;
      mrow[i]  = mn;
      alpha[i] = al;
      p0[i] = e0;
      p1[i] = e1;
    }

    // Rescale output accumulators
#pragma unroll
    for (int t = 0; t < 4; ++t)
#pragma unroll
      for (int i = 0; i < 8; ++i) o[t][i] *= alpha[i];

    // Re-stripe P from C-layout to A-fragment layout through LDS
#pragma unroll
    for (int i = 0; i < 8; ++i) {
      sP[wave][rowoff + i][colL]      = (_Float16)p0[i];
      sP[wave][rowoff + i][16 + colL] = (_Float16)p1[i];
    }
    __syncthreads();
    v16h pa;
    {
      int row = lane & 15;
      int kbv = (lane < 16) ? 0 : 8;
      const _Float16* pp = &sP[wave][row][kbv];
      v8h lo = *(const v8h*)pp;
      v8h hi = *(const v8h*)(pp + 16);
      pa = __builtin_shufflevector(lo, hi, 0, 1, 2, 3, 4, 5, 6, 7,
                                   8, 9, 10, 11, 12, 13, 14, 15);
    }
    __syncthreads();

    // O += P * V   (K = 32 keys; V^T rows are contiguous in key)
#pragma unroll
    for (int t = 0; t < 4; ++t) {
      v16h vb = load_b_frag(vh, n, t * 16, j0);
      o[t] = WMMA_F16(pa, vb, o[t]);
    }
  }

  // Normalize and store merged-head f16 output [b, n, d]
  float inv[8];
#pragma unroll
  for (int i = 0; i < 8; ++i) inv[i] = 1.0f / lrow[i];
#pragma unroll
  for (int t = 0; t < 4; ++t) {
#pragma unroll
    for (int i = 0; i < 8; ++i) {
      int qrow = qbase + rowoff + i;
      int c    = h * 64 + t * 16 + colL;
      out[((size_t)b * 2048 + qrow) * 1024 + c] = (_Float16)(o[t][i] * inv[i]);
    }
  }
}

// ---------------------------------------------------------------------------
// Launch: x[2,2048,1024] f32, Wq[1024,1024], Wkv[2048,1024], Wout[1024,1024]
// out: [2,2048,1024] f32
// ---------------------------------------------------------------------------
extern "C" void kernel_launch(void* const* d_in, const int* in_sizes, int n_in,
                              void* d_out, int out_size, void* d_ws, size_t ws_size,
                              hipStream_t stream) {
  const float* x    = (const float*)d_in[0];
  const float* Wq   = (const float*)d_in[1];
  const float* Wkv  = (const float*)d_in[2];
  const float* Wout = (const float*)d_in[3];
  float* outF = (float*)d_out;

  const int B = 2, N = 2048, D = 1024;
  const size_t xe = (size_t)B * N * D;  // 4M elements

  char* ws = (char*)d_ws;
  size_t off = 0;
  _Float16* xh    = (_Float16*)(ws + off); off += xe * 2;               // 8MB
  _Float16* wqh   = (_Float16*)(ws + off); off += (size_t)D * D * 2;    // 2MB
  _Float16* wkvh  = (_Float16*)(ws + off); off += (size_t)2 * D * D * 2;// 4MB
  _Float16* wouth = (_Float16*)(ws + off); off += (size_t)D * D * 2;    // 2MB
  _Float16* qb    = (_Float16*)(ws + off); off += xe * 2;               // 8MB
  _Float16* kbuf  = (_Float16*)(ws + off); off += xe * 2;               // 8MB
  _Float16* vbuf  = (_Float16*)(ws + off); off += xe * 2;               // 8MB
  _Float16* abuf  = (_Float16*)(ws + off); off += xe * 2;               // 8MB

  // f32 -> f16 staging
  cvt_f32_to_f16_kernel<<<(int)((xe + 255) / 256), 256, 0, stream>>>(x, xh, (int)xe);
  cvt_f32_to_f16_kernel<<<(D * D + 255) / 256, 256, 0, stream>>>(Wq, wqh, D * D);
  cvt_f32_to_f16_kernel<<<(2 * D * D + 255) / 256, 256, 0, stream>>>(Wkv, wkvh, 2 * D * D);
  cvt_f32_to_f16_kernel<<<(D * D + 255) / 256, 256, 0, stream>>>(Wout, wouth, D * D);

  // q = (x Wq^T) * dh^-0.5  ->  f16 [b,h,n,dh]
  wmma_gemm_kernel<<<dim3(64, 4), 256, 0, stream>>>(
      xh, wqh, B * N, D, D, /*mode=*/1, /*scale=*/0.125f, qb, qb, D, nullptr);
  // kv = x Wkv^T : k -> [b,h,n,dh], v -> transposed [b,h,dh,n]
  wmma_gemm_kernel<<<dim3(64, 8), 256, 0, stream>>>(
      xh, wkvh, B * N, 2 * D, D, /*mode=*/1, /*scale=*/1.0f, kbuf, vbuf, D, nullptr);
  // causal attention
  attn_kernel<<<dim3(16, 16, 2), 256, 0, stream>>>(qb, kbuf, vbuf, abuf);
  // out = attn Wout^T  (f32)
  wmma_gemm_kernel<<<dim3(64, 4), 256, 0, stream>>>(
      abuf, wouth, B * N, D, D, /*mode=*/0, /*scale=*/1.0f, nullptr, nullptr, 0, outF);
}
